// AdaptiveSelfReference_54614804136388
// MI455X (gfx1250) — compile-verified, matrix operand order checked by reading, not run on hardware
//
#include <hip/hip_runtime.h>
#include <stdint.h>

// ---------------- CDNA5 WMMA types ----------------
typedef __attribute__((ext_vector_type(16))) __bf16 v16bf;
typedef __attribute__((ext_vector_type(8)))  float  v8f;
typedef __attribute__((ext_vector_type(2)))  float  f2v;

union Frag {          // one wave-striped 16-bit matrix operand (8 VGPRs/lane)
  v16bf bf;
  uint4 q[2];
};
union Pack8 { uint16_t h[8]; uint4 q; };

__device__ __forceinline__ uint16_t bf16bits(float x) {
  union { float f; uint32_t u; } c; c.f = x;
  uint32_t r = c.u + 0x7FFFu + ((c.u >> 16) & 1u);   // round-to-nearest-even
  return (uint16_t)(r >> 16);
}
__device__ __forceinline__ v8f vzero8() {
  v8f z = {0.f,0.f,0.f,0.f,0.f,0.f,0.f,0.f};
  return z;
}
__device__ __forceinline__ v8f wmma_bf16(const Frag& a, const Frag& b, v8f c) {
  return __builtin_amdgcn_wmma_f32_16x16x32_bf16(false, a.bf, false, b.bf,
                                                 (short)0, c, false, false);
}

// A-fragment (16x32 bf16) from an LDS row-major tile.
// Layout (ISA 7.12.2): lanes 0-15 row M=lane, K = kb..kb+7 & kb+16..kb+23 with kb=0;
// lanes 16-31 same rows, kb=8.  -> two 16B contiguous LDS reads per lane.
__device__ __forceinline__ Frag load_a_lds(const uint16_t* base, int rowStride,
                                           int lane, int kt) {
  Frag f;
  const int m  = lane & 15;
  const int kb = kt * 32 + ((lane >> 4) << 3);
  const uint16_t* p = base + m * rowStride + kb;
  f.q[0] = *(const uint4*)(p);
  f.q[1] = *(const uint4*)(p + 16);
  return f;
}

// B-fragment from fragment-major prepacked weights: each lane reads its own
// contiguous 32 bytes -> two global b128 loads, fully coalesced per wave.
__device__ __forceinline__ Frag load_b_frag(const uint16_t* fw, int NT,
                                            int kt, int nt, int lane) {
  const uint16_t* p = fw + (((size_t)(kt * NT + nt) * 32 + lane) << 4);
  Frag f;
  f.q[0] = ((const uint4*)p)[0];
  f.q[1] = ((const uint4*)p)[1];
  return f;
}

// ---------------- prep: fp32 [K][N] -> bf16 fragment-major ----------------
__global__ __launch_bounds__(256) void prep_frag(const float* __restrict__ W,
                                                 uint16_t* __restrict__ dst,
                                                 int K, int N) {
  int idx = blockIdx.x * 256 + threadIdx.x;
  if (idx >= K * N) return;
  int j    = idx & 15;
  int lane = (idx >> 4) & 31;
  int fr   = idx >> 9;
  int NT = N >> 4;
  int nt = fr % NT, kt = fr / NT;
  int n = nt * 16 + (lane & 15);
  int k = kt * 32 + ((lane >> 4) << 3) + (j < 8 ? j : j + 8);
  dst[idx] = bf16bits(W[(size_t)k * N + n]);
}

// ---------------- LDS layout (padded strides, 319488 B total) ----------------
#define XSTR 520     // 512 + 8 pad (bank-conflict-free b128 reads)
#define HSTR 136     // 128 + 8
#define WSTR 72      // 64 + 8
#define SM_SW_OFF (8 * 16 * XSTR * 2)
#define SM_HW_OFF (SM_SW_OFF + 8 * 16 * XSTR * 2)
#define SM_WW_OFF (SM_HW_OFF + 8 * 16 * HSTR * 2)
#define SMEM_BYTES (SM_WW_OFF + 8 * 16 * WSTR * 2)   // 319488 <= 320KB/WGP

// ---------------- stage 1: per-row pipeline, 16 rows per wave ----------------
__global__ __launch_bounds__(256) void asr_main(
    const float* __restrict__ X,  const float* __restrict__ b1,
    const float* __restrict__ b2, const float* __restrict__ bu,
    const uint16_t* __restrict__ fw1, const uint16_t* __restrict__ fw2,
    const uint16_t* __restrict__ fwu, const uint16_t* __restrict__ fref,
    float* __restrict__ outSel, uint16_t* __restrict__ wgtT,
    uint16_t* __restrict__ updT, int Bn)
{
  extern __shared__ char smem[];
  const int wave = threadIdx.x >> 5;
  const int lane = threadIdx.x & 31;
  const int hl   = lane >> 4;      // which 16-lane half
  const int ln   = lane & 15;
  uint16_t* Xw = (uint16_t*)(smem)             + wave * (16 * XSTR);
  uint16_t* Sw = (uint16_t*)(smem + SM_SW_OFF) + wave * (16 * XSTR);
  uint16_t* Hw = (uint16_t*)(smem + SM_HW_OFF) + wave * (16 * HSTR);
  uint16_t* Ww = (uint16_t*)(smem + SM_WW_OFF) + wave * (16 * WSTR);
  const int r0 = blockIdx.x * 128 + wave * 16;   // this wave's 16 rows

  // ---- stage experience tile into LDS as bf16 (streaming, non-temporal) ----
  {
    const f2v* src = (const f2v*)(X + (size_t)r0 * 512);
    uint32_t* dst = (uint32_t*)Xw;
    for (int p = lane; p < 16 * 256; p += 32) {
      int row = p >> 8, cp = p & 255;
      f2v v = __builtin_nontemporal_load(src + p);
      dst[row * (XSTR / 2) + cp] =
          (uint32_t)bf16bits(v.x) | ((uint32_t)bf16bits(v.y) << 16);
    }
  }
  __syncthreads();

  // ---- GEMM1: H[16,128] = X[16,512] @ w1 ; relu ----
  v8f acc1[8];
  #pragma unroll
  for (int nt = 0; nt < 8; ++nt) acc1[nt] = vzero8();
  for (int kt = 0; kt < 16; ++kt) {
    Frag a = load_a_lds(Xw, XSTR, lane, kt);
    Frag b[8];
    #pragma unroll
    for (int nt = 0; nt < 8; ++nt) b[nt] = load_b_frag(fw1, 8, kt, nt, lane);
    #pragma unroll
    for (int nt = 0; nt < 8; ++nt) acc1[nt] = wmma_bf16(a, b[nt], acc1[nt]);
  }
  #pragma unroll
  for (int nt = 0; nt < 8; ++nt) {
    float bb = b1[nt * 16 + ln];
    #pragma unroll
    for (int r = 0; r < 8; ++r) {
      float v = fmaxf(acc1[nt][r] + bb, 0.0f);
      Hw[(r + 8 * hl) * HSTR + nt * 16 + ln] = bf16bits(v);
    }
  }
  __syncthreads();

  // ---- GEMM2: logits[16,64] = H @ w2 ; softmax over 64 ----
  v8f acc2[4];
  #pragma unroll
  for (int nt = 0; nt < 4; ++nt) acc2[nt] = vzero8();
  for (int kt = 0; kt < 4; ++kt) {
    Frag a = load_a_lds(Hw, HSTR, lane, kt);
    Frag b[4];
    #pragma unroll
    for (int nt = 0; nt < 4; ++nt) b[nt] = load_b_frag(fw2, 4, kt, nt, lane);
    #pragma unroll
    for (int nt = 0; nt < 4; ++nt) acc2[nt] = wmma_bf16(a, b[nt], acc2[nt]);
  }
  #pragma unroll
  for (int nt = 0; nt < 4; ++nt) {
    float bb = b2[nt * 16 + ln];
    #pragma unroll
    for (int r = 0; r < 8; ++r) acc2[nt][r] += bb;
  }
  // row max / sum: N is striped over 16 lanes of each half + 4 accumulators
  float mx[8], sm[8];
  #pragma unroll
  for (int r = 0; r < 8; ++r) {
    float m = fmaxf(fmaxf(acc2[0][r], acc2[1][r]), fmaxf(acc2[2][r], acc2[3][r]));
    #pragma unroll
    for (int off = 1; off < 16; off <<= 1) m = fmaxf(m, __shfl_xor(m, off, 32));
    mx[r] = m;
  }
  #pragma unroll
  for (int nt = 0; nt < 4; ++nt)
    #pragma unroll
    for (int r = 0; r < 8; ++r) acc2[nt][r] = __expf(acc2[nt][r] - mx[r]);
  #pragma unroll
  for (int r = 0; r < 8; ++r) {
    float s = acc2[0][r] + acc2[1][r] + acc2[2][r] + acc2[3][r];
    #pragma unroll
    for (int off = 1; off < 16; off <<= 1) s += __shfl_xor(s, off, 32);
    sm[r] = 1.0f / s;
  }
  // store softmax weights: LDS (A-operand of GEMM3) + global transposed for stage 2
  #pragma unroll
  for (int nt = 0; nt < 4; ++nt) {
    Pack8 pk;
    #pragma unroll
    for (int r = 0; r < 8; ++r) {
      uint16_t h = bf16bits(acc2[nt][r] * sm[r]);
      Ww[(r + 8 * hl) * WSTR + nt * 16 + ln] = h;
      pk.h[r] = h;
    }
    int s = nt * 16 + ln;
    *(uint4*)(wgtT + (size_t)s * Bn + r0 + 8 * hl) = pk.q;   // wgtT[s][k0..k0+7]
  }
  __syncthreads();

  // ---- GEMM3: sel[16,512] = wgt[16,64] @ ref ; write fp32 out + bf16 LDS ----
  for (int ng = 0; ng < 4; ++ng) {
    v8f acc[8];
    #pragma unroll
    for (int j = 0; j < 8; ++j) acc[j] = vzero8();
    #pragma unroll
    for (int kt = 0; kt < 2; ++kt) {
      Frag a = load_a_lds(Ww, WSTR, lane, kt);
      Frag b[8];
      #pragma unroll
      for (int j = 0; j < 8; ++j) b[j] = load_b_frag(fref, 32, kt, ng * 8 + j, lane);
      #pragma unroll
      for (int j = 0; j < 8; ++j) acc[j] = wmma_bf16(a, b[j], acc[j]);
    }
    #pragma unroll
    for (int j = 0; j < 8; ++j) {
      int n = (ng * 8 + j) * 16 + ln;
      #pragma unroll
      for (int r = 0; r < 8; ++r) {
        int row = r + 8 * hl;
        __builtin_nontemporal_store(acc[j][r],
                                    outSel + (size_t)(r0 + row) * 512 + n);
        Sw[row * XSTR + n] = bf16bits(acc[j][r]);
      }
    }
  }
  __syncthreads();

  // ---- GEMM4: upd = tanh(X@wu_top + sel@wu_bot + bu); write updT (bf16) ----
  for (int ng = 0; ng < 4; ++ng) {
    v8f acc[8];
    #pragma unroll
    for (int j = 0; j < 8; ++j) acc[j] = vzero8();
    for (int kt = 0; kt < 32; ++kt) {   // K=1024: [X | sel]
      Frag a = (kt < 16) ? load_a_lds(Xw, XSTR, lane, kt)
                         : load_a_lds(Sw, XSTR, lane, kt - 16);
      Frag b[8];
      #pragma unroll
      for (int j = 0; j < 8; ++j) b[j] = load_b_frag(fwu, 32, kt, ng * 8 + j, lane);
      #pragma unroll
      for (int j = 0; j < 8; ++j) acc[j] = wmma_bf16(a, b[j], acc[j]);
    }
    #pragma unroll
    for (int j = 0; j < 8; ++j) {
      int n = (ng * 8 + j) * 16 + ln;
      float bb = bu[n];
      Pack8 pk;
      #pragma unroll
      for (int r = 0; r < 8; ++r) pk.h[r] = bf16bits(tanhf(acc[j][r] + bb));
      *(uint4*)(updT + (size_t)n * Bn + r0 + 8 * hl) = pk.q;  // updT[n][k0..k0+7]
    }
  }
}

// ---- stage 2: new_ref[64,512] = ref + 0.01 * wgtT[64,B] @ upd[B,512] ----
// 128 output tiles, one wave each. K=131072 split over 4 independent WMMA
// chains so the loop runs at WMMA issue rate, not latency. No atomics:
// every output element has a single owner -> bit-deterministic.
__global__ __launch_bounds__(256) void asr_refupd(
    const uint16_t* __restrict__ wgtT, const uint16_t* __restrict__ updT,
    const float* __restrict__ ref, float* __restrict__ outRef, int Bn)
{
  const int wave = threadIdx.x >> 5, lane = threadIdx.x & 31;
  const int hl = lane >> 4, ln = lane & 15;
  const int tile = blockIdx.x * 8 + wave;   // 0..127
  const int mt = tile >> 5;                 // S tile  (0..3)
  const int nt = tile & 31;                 // D tile  (0..31)
  const int KT = Bn >> 5;
  const uint16_t* arow = wgtT + (size_t)(mt * 16 + ln) * Bn;
  const uint16_t* brow = updT + (size_t)(nt * 16 + ln) * Bn;
  v8f acc[4];
  #pragma unroll
  for (int u = 0; u < 4; ++u) acc[u] = vzero8();
  for (int kt = 0; kt < KT; kt += 4) {
    Frag a[4], b[4];
    #pragma unroll
    for (int u = 0; u < 4; ++u) {
      int kb = (kt + u) * 32 + 8 * hl;
      a[u].q[0] = *(const uint4*)(arow + kb);
      a[u].q[1] = *(const uint4*)(arow + kb + 16);
      b[u].q[0] = *(const uint4*)(brow + kb);
      b[u].q[1] = *(const uint4*)(brow + kb + 16);
    }
    #pragma unroll
    for (int u = 0; u < 4; ++u) acc[u] = wmma_bf16(a[u], b[u], acc[u]);
  }
  v8f tot = (acc[0] + acc[1]) + (acc[2] + acc[3]);
  #pragma unroll
  for (int r = 0; r < 8; ++r) {
    int s = mt * 16 + r + 8 * hl;
    int n = nt * 16 + ln;
    outRef[s * 512 + n] = ref[s * 512 + n] + 0.01f * tot[r];
  }
}

// ---------------------------------------------------------------------------
extern "C" void kernel_launch(void* const* d_in, const int* in_sizes, int n_in,
                              void* d_out, int out_size, void* d_ws, size_t ws_size,
                              hipStream_t stream) {
  (void)n_in; (void)out_size; (void)ws_size;
  const float* X   = (const float*)d_in[0];   // [B,512]
  const float* ref = (const float*)d_in[1];   // [64,512]
  const float* w1  = (const float*)d_in[2];   // [512,128]
  const float* b1  = (const float*)d_in[3];
  const float* w2  = (const float*)d_in[4];   // [128,64]
  const float* b2  = (const float*)d_in[5];
  const float* wu  = (const float*)d_in[6];   // [1024,512]
  const float* bu  = (const float*)d_in[7];
  const int Bn = in_sizes[0] / 512;

  float* outSel = (float*)d_out;
  float* outRef = outSel + (size_t)Bn * 512;

  size_t off = 0;
  auto carve = [&](size_t bytes) -> uint16_t* {
    uint16_t* p = (uint16_t*)((char*)d_ws + off);
    off = (off + bytes + 255) & ~(size_t)255;
    return p;
  };
  uint16_t* fw1  = carve((size_t)512 * 128 * 2);
  uint16_t* fw2  = carve((size_t)128 * 64  * 2);
  uint16_t* fwu  = carve((size_t)1024 * 512 * 2);
  uint16_t* fref = carve((size_t)64 * 512 * 2);
  uint16_t* wgtT = carve((size_t)64  * Bn * 2);   // [64][B]  bf16
  uint16_t* updT = carve((size_t)512 * Bn * 2);   // [512][B] bf16

  prep_frag<<<(512 * 128 + 255) / 256, 256, 0, stream>>>(w1, fw1, 512, 128);
  prep_frag<<<(128 * 64  + 255) / 256, 256, 0, stream>>>(w2, fw2, 128, 64);
  prep_frag<<<(1024 * 512 + 255) / 256, 256, 0, stream>>>(wu, fwu, 1024, 512);
  prep_frag<<<(64 * 512 + 255) / 256, 256, 0, stream>>>(ref, fref, 64, 512);

  hipFuncSetAttribute(reinterpret_cast<const void*>(asr_main),
                      hipFuncAttributeMaxDynamicSharedMemorySize, SMEM_BYTES);
  asr_main<<<Bn / 128, 256, SMEM_BYTES, stream>>>(
      X, b1, b2, bu, fw1, fw2, fwu, fref, outSel, wgtT, updT, Bn);
  asr_refupd<<<16, 256, 0, stream>>>(wgtT, updT, ref, outRef, Bn);
}
